// TimeAttention_1425929142989
// MI455X (gfx1250) — matrix operand order repaired and checked by reading, not compiled, not run on hardware
//
#include <hip/hip_runtime.h>
#include <hip/hip_bf16.h>

// ---------------------------------------------------------------------------
// TimeSformer divided time-attention for MI455X (gfx1250, wave32, WMMA).
// GEMMs + attention contractions run through v_wmma_f32_16x16x32_bf16.
// Kernel 4 stages its bf16 A-tile with GLOBAL_LOAD_ASYNC_TO_LDS_B128
// (ASYNCcnt path) when the toolchain exposes the builtin, else falls back.
// ---------------------------------------------------------------------------

typedef __attribute__((ext_vector_type(16))) __bf16 v16bf;
typedef __attribute__((ext_vector_type(8)))  __bf16 v8bf;
typedef __attribute__((ext_vector_type(4)))  __bf16 v4bf;
typedef __attribute__((ext_vector_type(8)))  float  v8f;
typedef __attribute__((ext_vector_type(4)))  float  v4f;
typedef __attribute__((ext_vector_type(4)))  int    v4i;

#define NTOK      3137          // 1 + f*n
#define TOK_TOTAL 25096         // B * NTOK
#define DIM       512
#define QKVC      1536
#define HN        8
#define DH        64
#define BH        64            // B * H
#define SP        196           // spatial positions n

static __device__ __forceinline__ v16bf make_frag(v8bf lo, v8bf hi) {
    union { v16bf v; v8bf h[2]; } u;
    u.h[0] = lo; u.h[1] = hi;
    return u.v;
}

static __device__ __forceinline__ v4bf cvt4(v4f v) {
    v4bf r;
    r[0] = (__bf16)v[0]; r[1] = (__bf16)v[1];
    r[2] = (__bf16)v[2]; r[3] = (__bf16)v[3];
    return r;
}

#if defined(__AMDGCN__) && __has_builtin(__builtin_amdgcn_global_load_async_to_lds_b128)
#define HAVE_ASYNC_LDS 1
typedef __attribute__((address_space(1))) v4i* g1_v4i;   // global int4*
typedef __attribute__((address_space(3))) v4i* l3_v4i;   // LDS int4*
static __device__ __forceinline__ void async_copy16(const void* g, void* l) {
    __builtin_amdgcn_global_load_async_to_lds_b128((g1_v4i)g, (l3_v4i)l, 0, 0);
}
static __device__ __forceinline__ void async_wait0() {
#if __has_builtin(__builtin_amdgcn_s_wait_asynccnt)
    __builtin_amdgcn_s_wait_asynccnt(0);
#else
    asm volatile("s_wait_asynccnt 0" ::: "memory");
#endif
}
#else
#define HAVE_ASYNC_LDS 0
#endif

// ---------------------------------------------------------------------------
// Kernel 1: qkv = x @ W_qkv   (fp32 in, bf16 staged, WMMA bf16, fp32 accum)
// Output scattered into q/k/v buffers laid out (B*H, NTOK, DH); q pre-scaled.
// ---------------------------------------------------------------------------
__global__ __launch_bounds__(128) void qkv_gemm_kernel(
    const float* __restrict__ x, const float* __restrict__ Wqkv,
    __bf16* __restrict__ qb, __bf16* __restrict__ kbuf, __bf16* __restrict__ vbuf)
{
    __shared__ __bf16 Abuf[64][32];
    __shared__ __bf16 Bt[64][32];     // B transposed: Bt[n][k]

    const int tid  = threadIdx.x;
    const int wave = tid >> 5;
    const int lane = tid & 31;
    const int m0   = blockIdx.x * 64;
    const int n0   = blockIdx.y * 64;

    v8f acc[4] = {};

    for (int k0 = 0; k0 < DIM; k0 += 32) {
        // stage A tile (64 x 32) fp32 -> bf16, vectorized b128 loads / b64 stores
#pragma unroll
        for (int idx = tid; idx < 512; idx += 128) {
            int r = idx >> 3, c = (idx & 7) * 4;
            int row = m0 + r;
            v4f v = {};
            if (row < TOK_TOTAL) {
                v = *(const v4f*)&x[(long)row * DIM + (k0 + c)];
                if (k0 + 32 < DIM)
                    __builtin_prefetch(&x[(long)row * DIM + (k0 + 32 + c)], 0, 3);
            }
            *(v4bf*)&Abuf[r][c] = cvt4(v);
        }
        // stage B tile transposed (W_qkv is [K][N] row-major)
#pragma unroll
        for (int idx = tid; idx < 512; idx += 128) {
            int kk = idx >> 4, nn = (idx & 15) * 4;
            v4f w = *(const v4f*)&Wqkv[(long)(k0 + kk) * QKVC + (n0 + nn)];
            Bt[nn + 0][kk] = (__bf16)w[0];
            Bt[nn + 1][kk] = (__bf16)w[1];
            Bt[nn + 2][kk] = (__bf16)w[2];
            Bt[nn + 3][kk] = (__bf16)w[3];
        }
        __syncthreads();

        const int am = wave * 16 + (lane & 15);
        const int kb = (lane >> 4) * 8;
        v16bf a = make_frag(*(const v8bf*)&Abuf[am][kb],
                            *(const v8bf*)&Abuf[am][16 + kb]);
#pragma unroll
        for (int ct = 0; ct < 4; ++ct) {
            const int bn = ct * 16 + (lane & 15);
            const int bk = (lane >> 4) * 8;
            v16bf b = make_frag(*(const v8bf*)&Bt[bn][bk],
                                *(const v8bf*)&Bt[bn][bk + 16]);
            acc[ct] = __builtin_amdgcn_wmma_f32_16x16x32_bf16(
                false, a, false, b, (short)0, acc[ct], false, false);
        }
        __syncthreads();
    }

    // scatter C into q/k/v head-major buffers
#pragma unroll
    for (int ct = 0; ct < 4; ++ct) {
#pragma unroll
        for (int r = 0; r < 8; ++r) {
            int row = m0 + wave * 16 + (lane >> 4) * 8 + r;
            if (row >= TOK_TOTAL) continue;
            int c = n0 + ct * 16 + (lane & 15);
            int b = row / NTOK, t = row - b * NTOK;
            float v = acc[ct][r];
            if (c < DIM) {
                int hh = c >> 6, d = c & 63;
                qb[((long)(b * HN + hh) * NTOK + t) * DH + d] = (__bf16)(v * 0.125f);
            } else if (c < 2 * DIM) {
                int cc = c - DIM, hh = cc >> 6, d = cc & 63;
                kbuf[((long)(b * HN + hh) * NTOK + t) * DH + d] = (__bf16)v;
            } else {
                int cc = c - 2 * DIM, hh = cc >> 6, d = cc & 63;
                vbuf[((long)(b * HN + hh) * NTOK + t) * DH + d] = (__bf16)v;
            }
        }
    }
}

// ---------------------------------------------------------------------------
// Kernel 2: cls-token attention. 1 block per (b,h); q[0] attends all 3137 keys.
// ---------------------------------------------------------------------------
__global__ __launch_bounds__(256) void cls_attn_kernel(
    const __bf16* __restrict__ qb, const __bf16* __restrict__ kbuf,
    const __bf16* __restrict__ vbuf, __bf16* __restrict__ attn)
{
    __shared__ float qv[DH];
    __shared__ float s[NTOK];
    __shared__ float red[256];

    const int bh  = blockIdx.x;
    const int tid = threadIdx.x;
    const long base = (long)bh * NTOK * DH;

    if (tid < DH) qv[tid] = (float)qb[base + tid];   // q pre-scaled
    __syncthreads();

    float lmax = -1e30f;
    for (int j = tid; j < NTOK; j += 256) {
        const v8bf* kr8 = (const v8bf*)(kbuf + base + (long)j * DH);
        float dot = 0.f;
#pragma unroll
        for (int i = 0; i < 8; ++i) {
            v8bf kv = kr8[i];
#pragma unroll
            for (int e = 0; e < 8; ++e) dot += qv[i * 8 + e] * (float)kv[e];
        }
        s[j] = dot;
        lmax = fmaxf(lmax, dot);
    }
    red[tid] = lmax; __syncthreads();
    for (int off = 128; off > 0; off >>= 1) {
        if (tid < off) red[tid] = fmaxf(red[tid], red[tid + off]);
        __syncthreads();
    }
    const float m = red[0]; __syncthreads();

    float lsum = 0.f;
    for (int j = tid; j < NTOK; j += 256) {
        float e = __expf(s[j] - m);
        s[j] = e; lsum += e;
    }
    red[tid] = lsum; __syncthreads();
    for (int off = 128; off > 0; off >>= 1) {
        if (tid < off) red[tid] += red[tid + off];
        __syncthreads();
    }
    const float denom = red[0]; __syncthreads();

    const int d = tid & 63, sid = tid >> 6;
    float a = 0.f;
    for (int j = sid; j < NTOK; j += 4)
        a += s[j] * (float)vbuf[base + (long)j * DH + d];
    red[tid] = a; __syncthreads();
    if (tid < 64) {
        float tot = red[tid] + red[tid + 64] + red[tid + 128] + red[tid + 192];
        int b = bh >> 3, h = bh & 7;
        attn[(long)b * NTOK * (HN * DH) + h * DH + tid] = (__bf16)(tot / denom);
    }
}

// ---------------------------------------------------------------------------
// Kernel 3: time attention. 1 wave per (bh, spatial) group; 8 groups / block.
// Q=(16x64), K/V=(17x64, cls-prepended). 8 WMMAs per group.
// ---------------------------------------------------------------------------
__global__ __launch_bounds__(256) void time_attn_kernel(
    const __bf16* __restrict__ qb, const __bf16* __restrict__ kbuf,
    const __bf16* __restrict__ vbuf, __bf16* __restrict__ attn)
{
    __shared__ float  Ssm[8][16][32];
    __shared__ __bf16 Pb[8][16][32];
    __shared__ __bf16 Vt[8][64][32];  // Vt[d][j], j padded to 32

    const int tid  = threadIdx.x;
    const int wave = tid >> 5;
    const int lane = tid & 31;
    const int g    = blockIdx.x * 8 + wave;       // 0..12543 exactly
    const int bh   = g / SP;
    const int sp   = g - bh * SP;
    const long base = (long)bh * NTOK * DH;

    const int lm = lane & 15;   // A row m / B-C column n
    const int lh = lane >> 4;
    const int kbq = lh * 8;

    // ---- S = Q K^T : two K-dim steps of 32, two column tiles (j 0-15, j 16)
    v8f s0 = {}, s1 = {};
#pragma unroll
    for (int st = 0; st < 2; ++st) {
        const long qoff = base + (long)(1 + lm * SP + sp) * DH + st * 32;
        v16bf a = make_frag(*(const v8bf*)(qb + qoff + kbq),
                            *(const v8bf*)(qb + qoff + 16 + kbq));
        {   // keys j = 0..15  (j==0 is the cls key)
            int j = lm;
            long tok = (j == 0) ? 0 : (long)(1 + (j - 1) * SP + sp);
            long koff = base + tok * DH + st * 32 + lh * 8;
            v16bf b = make_frag(*(const v8bf*)(kbuf + koff),
                                *(const v8bf*)(kbuf + koff + 16));
            s0 = __builtin_amdgcn_wmma_f32_16x16x32_bf16(
                false, a, false, b, (short)0, s0, false, false);
        }
        {   // key j = 16 (frame 15) in column 0; other columns zero
            v8bf z = {};
            v16bf b = make_frag(z, z);
            if (lm == 0) {
                long koff = base + (long)(1 + 15 * SP + sp) * DH + st * 32 + lh * 8;
                b = make_frag(*(const v8bf*)(kbuf + koff),
                              *(const v8bf*)(kbuf + koff + 16));
            }
            s1 = __builtin_amdgcn_wmma_f32_16x16x32_bf16(
                false, a, false, b, (short)0, s1, false, false);
        }
    }

    // dump S to LDS (lane holds column lm, rows lh*8+r)
#pragma unroll
    for (int r = 0; r < 8; ++r) {
        Ssm[wave][lh * 8 + r][lm]      = s0[r];
        Ssm[wave][lh * 8 + r][16 + lm] = s1[r];
    }
    __syncthreads();

    // ---- softmax over 17 keys (lanes 0..15: one row each), write P (bf16)
    if (lane < 16) {
        float mx = -1e30f;
#pragma unroll
        for (int j = 0; j <= 16; ++j) mx = fmaxf(mx, Ssm[wave][lane][j]);
        float sum = 0.f;
#pragma unroll
        for (int j = 0; j <= 16; ++j) {
            float e = __expf(Ssm[wave][lane][j] - mx);
            sum += e;
            Ssm[wave][lane][j] = e;
        }
        float inv = 1.0f / sum;
#pragma unroll
        for (int j = 0; j < 32; ++j)
            Pb[wave][lane][j] = (__bf16)((j <= 16) ? Ssm[wave][lane][j] * inv : 0.f);
    }

    // ---- stage V transposed: Vt[d][j] (zero-padded j=17..31)
    {
        unsigned int* vz = (unsigned int*)&Vt[wave][0][0];
#pragma unroll
        for (int i = 0; i < 32; ++i) vz[lane + 32 * i] = 0u;
        for (int j = 0; j <= 16; ++j) {
            long tok = (j == 0) ? 0 : (long)(1 + (j - 1) * SP + sp);
            long voff = base + tok * DH + lane * 2;
            Vt[wave][lane * 2][j]     = vbuf[voff];
            Vt[wave][lane * 2 + 1][j] = vbuf[voff + 1];
        }
    }
    __syncthreads();

    // ---- O = P V : K-dim = 32 (padded keys), 4 column tiles of DH
    v16bf pa = make_frag(*(const v8bf*)&Pb[wave][lm][kbq],
                         *(const v8bf*)&Pb[wave][lm][16 + kbq]);
    v8f o[4] = {};
#pragma unroll
    for (int ct = 0; ct < 4; ++ct) {
        const int d = ct * 16 + lm;
        v16bf b = make_frag(*(const v8bf*)&Vt[wave][d][lh * 8],
                            *(const v8bf*)&Vt[wave][d][lh * 8 + 16]);
        o[ct] = __builtin_amdgcn_wmma_f32_16x16x32_bf16(
            false, pa, false, b, (short)0, o[ct], false, false);
    }

    // ---- store to attn (B, NTOK, H*DH) in bf16
    const int b_ = bh >> 3, h_ = bh & 7;
#pragma unroll
    for (int ct = 0; ct < 4; ++ct) {
#pragma unroll
        for (int r = 0; r < 8; ++r) {
            int m   = lh * 8 + r;
            int tok = 1 + m * SP + sp;
            int d   = ct * 16 + lm;
            attn[((long)b_ * NTOK + tok) * (HN * DH) + h_ * DH + d] = (__bf16)o[ct][r];
        }
    }
}

// ---------------------------------------------------------------------------
// Kernel 4: out = attn @ W_out + b_out  (bf16 A via async LDS DMA, fp32 out)
// ---------------------------------------------------------------------------
__global__ __launch_bounds__(128) void out_proj_kernel(
    const __bf16* __restrict__ attn, const float* __restrict__ Wout,
    const float* __restrict__ bias, float* __restrict__ out)
{
    __shared__ __bf16 Abuf[64][32];
    __shared__ __bf16 Bt[64][32];

    const int tid  = threadIdx.x;
    const int wave = tid >> 5;
    const int lane = tid & 31;
    const int m0   = blockIdx.x * 64;
    const int n0   = blockIdx.y * 64;

    v8f acc[4] = {};

    for (int k0 = 0; k0 < DIM; k0 += 32) {
        // stage A tile (64x32 bf16 = 256 x 16B chunks): async global->LDS DMA
#pragma unroll
        for (int idx = tid; idx < 256; idx += 128) {
            int r = idx >> 2, c = (idx & 3) * 8;
            int row = m0 + r;
            if (row < TOK_TOTAL) {
#if HAVE_ASYNC_LDS
                async_copy16(attn + (long)row * DIM + (k0 + c), &Abuf[r][c]);
#else
                *(v8bf*)&Abuf[r][c] =
                    *(const v8bf*)&attn[(long)row * DIM + (k0 + c)];
#endif
            } else {
                v8bf z = {};
                *(v8bf*)&Abuf[r][c] = z;
            }
        }
        // stage B tile transposed (W_out is [K][N] row-major fp32)
#pragma unroll
        for (int idx = tid; idx < 512; idx += 128) {
            int kk = idx >> 4, nn = (idx & 15) * 4;
            v4f w = *(const v4f*)&Wout[(long)(k0 + kk) * DIM + (n0 + nn)];
            Bt[nn + 0][kk] = (__bf16)w[0];
            Bt[nn + 1][kk] = (__bf16)w[1];
            Bt[nn + 2][kk] = (__bf16)w[2];
            Bt[nn + 3][kk] = (__bf16)w[3];
        }
#if HAVE_ASYNC_LDS
        async_wait0();
#endif
        __syncthreads();

        const int am = wave * 16 + (lane & 15);
        const int kb = (lane >> 4) * 8;
        v16bf a = make_frag(*(const v8bf*)&Abuf[am][kb],
                            *(const v8bf*)&Abuf[am][16 + kb]);
#pragma unroll
        for (int ct = 0; ct < 4; ++ct) {
            const int bn = ct * 16 + (lane & 15);
            const int bk = (lane >> 4) * 8;
            v16bf b = make_frag(*(const v8bf*)&Bt[bn][bk],
                                *(const v8bf*)&Bt[bn][bk + 16]);
            acc[ct] = __builtin_amdgcn_wmma_f32_16x16x32_bf16(
                false, a, false, b, (short)0, acc[ct], false, false);
        }
        __syncthreads();
    }

#pragma unroll
    for (int ct = 0; ct < 4; ++ct) {
#pragma unroll
        for (int r = 0; r < 8; ++r) {
            int row = m0 + wave * 16 + (lane >> 4) * 8 + r;
            if (row >= TOK_TOTAL) continue;
            int col = n0 + ct * 16 + (lane & 15);
            out[(long)row * DIM + col] = acc[ct][r] + bias[col];
        }
    }
}

// ---------------------------------------------------------------------------
extern "C" void kernel_launch(void* const* d_in, const int* in_sizes, int n_in,
                              void* d_out, int out_size, void* d_ws, size_t ws_size,
                              hipStream_t stream) {
    const float* x    = (const float*)d_in[0];
    const float* Wqkv = (const float*)d_in[1];
    const float* Wout = (const float*)d_in[2];
    const float* bout = (const float*)d_in[3];
    // d_in[4] = f, d_in[5] = n : compile-time constants here (16, 196)

    char* ws = (char*)d_ws;
    const size_t qsz = (size_t)BH * NTOK * DH * sizeof(__bf16);  // 25,698,304 B (16B aligned)
    __bf16* qb   = (__bf16*)(ws);
    __bf16* kb   = (__bf16*)(ws + qsz);
    __bf16* vb   = (__bf16*)(ws + 2 * qsz);
    __bf16* attn = (__bf16*)(ws + 3 * qsz);
    float*  out  = (float*)d_out;

    dim3 g1((TOK_TOTAL + 63) / 64, QKVC / 64);
    hipLaunchKernelGGL(qkv_gemm_kernel, g1, dim3(128), 0, stream, x, Wqkv, qb, kb, vb);

    hipLaunchKernelGGL(cls_attn_kernel, dim3(BH), dim3(256), 0, stream, qb, kb, vb, attn);

    hipLaunchKernelGGL(time_attn_kernel, dim3((BH * SP) / 8), dim3(256), 0, stream,
                       qb, kb, vb, attn);

    dim3 g4((TOK_TOTAL + 63) / 64, DIM / 64);
    hipLaunchKernelGGL(out_proj_kernel, g4, dim3(128), 0, stream, attn, Wout, bout, out);
}
